// FeatureExtractor_24369644437625
// MI455X (gfx1250) — matrix operand order repaired and checked by reading, not compiled, not run on hardware
//
#include <hip/hip_runtime.h>
#include <hip/hip_bf16.h>

// ---------------------------------------------------------------------------
// PointNet++ (SSG) forward for MI455X (gfx1250, wave32).
// FPS + kNN: latency-bound scalar/LDS kernels.
// All MLP matmuls: v_wmma_f32_16x16x32_f16, one wave per 16-row tile.
// Weights are pre-packed ONCE into B-fragment layout (f16, lane-major) so the
// hot loop does 2x16B loads per B operand instead of 16 scattered f32+cvt.
// ---------------------------------------------------------------------------

typedef __attribute__((ext_vector_type(16))) _Float16 v16h;
typedef __attribute__((ext_vector_type(8)))  _Float16 v8h;
typedef __attribute__((ext_vector_type(8)))  float    v8f;

#define DEVINL __device__ __forceinline__

// --- WMMA fragment helpers (layouts per CDNA5 ISA §7.12.2) -------

// A (16x32 f16): lane<16 -> rows M=lane, K = hi*8+i (i<8) / 16+hi*8+(i-8).
// Per lane this is two contiguous 8-half (16B) runs -> two vector loads.
DEVINL v16h load_a_frag(const _Float16* base, int lda) {
  int lane = threadIdx.x & 31;
  int row  = lane & 15;
  int hi   = lane >> 4;
  const _Float16* r0 = base + row * lda + hi * 8;
  v8h lo = *(const v8h*)r0;          // K = hi*8 .. hi*8+7
  v8h hp = *(const v8h*)(r0 + 16);   // K = 16+hi*8 .. 16+hi*8+7
  v16h a;
#pragma unroll
  for (int i = 0; i < 8; ++i) { a[i] = lo[i]; a[8 + i] = hp[i]; }
  return a;
}

// B fragment from pre-packed tile: 512 halves per 32x16 tile, lane-major
// (lane = hi*16+col owns halves [lane*16 .. lane*16+15] = K hi*8+i (i<8) / 16+hi*8+(i-8), N=col).
DEVINL v16h load_b_frag_packed(const _Float16* tile) {
  int lane = threadIdx.x & 31;
  const _Float16* src = tile + (size_t)lane * 16;
  v8h lo = *(const v8h*)src;
  v8h hp = *(const v8h*)(src + 8);
  v16h b;
#pragma unroll
  for (int i = 0; i < 8; ++i) { b[i] = lo[i]; b[8 + i] = hp[i]; }
  return b;
}

DEVINL v8f wmma_f16(v16h a, v16h b, v8f c) {
  return __builtin_amdgcn_wmma_f32_16x16x32_f16(
      /*neg_a=*/false, a, /*neg_b=*/false, b,
      /*c_mod=*/(short)0, c, /*reuse_a=*/false, /*reuse_b=*/false);
}

// --- Kernel W: pack W (cout x cin, f32, row-major) into B-frag tiles --------
// tile index = kt * NT + nt ; element e in [0,512): lane=e>>4 (col=lane&15,
// hi=lane>>4), i=e&15 -> K = kt*32 + (i<8 ? hi*8+i : 16+hi*8+(i-8)), N = nt*16 + col. Zero-pad k>=cin.
__global__ __launch_bounds__(256)
void pack_w_kernel(const float* __restrict__ W, _Float16* __restrict__ out,
                   int cin, int NT, int total) {
  int t = blockIdx.x * 256 + threadIdx.x;
  if (t >= total) return;
  int e    = t & 511;
  int tile = t >> 9;
  int nt = tile % NT, kt = tile / NT;
  int lane = e >> 4, i = e & 15;
  int col = lane & 15, hi = lane >> 4;
  int k = kt * 32 + ((i < 8) ? (hi * 8 + i) : (16 + hi * 8 + (i - 8)));
  int n = nt * 16 + col;
  float w = (k < cin) ? W[(size_t)n * cin + k] : 0.0f;
  out[t] = (_Float16)w;
}

// --- Kernel 0: (B,3,N) -> (B,N,3) point-major transpose ---------------------
__global__ __launch_bounds__(256)
void transpose_pc_kernel(const float* __restrict__ pc, float* __restrict__ xyz,
                         int B, int N) {
  int t = blockIdx.x * 256 + threadIdx.x;
  if (t >= B * N) return;
  int b = t / N, n = t % N;
#pragma unroll
  for (int c = 0; c < 3; ++c)
    xyz[((size_t)b * N + n) * 3 + c] = pc[((size_t)b * 3 + c) * N + n];
}

// --- Kernel 1: farthest-point sampling (one workgroup per batch) ------------
template<int N, int NPOINT>
__global__ __launch_bounds__(256)
void fps_kernel(const float* __restrict__ xyz, float* __restrict__ new_xyz) {
  constexpr int PTS = N / 256;
  int b = blockIdx.x;
  const float* X = xyz + (size_t)b * N * 3;
  float dist[PTS];
#pragma unroll
  for (int j = 0; j < PTS; ++j) dist[j] = 1e10f;

  __shared__ int   s_far;
  __shared__ float s_rv[256];
  __shared__ int   s_ri[256];
  if (threadIdx.x == 0) s_far = 0;
  __syncthreads();

  for (int it = 0; it < NPOINT; ++it) {
    int far = s_far;
    float cx = X[far * 3 + 0], cy = X[far * 3 + 1], cz = X[far * 3 + 2];
    if (threadIdx.x == 0) {
      new_xyz[((size_t)b * NPOINT + it) * 3 + 0] = cx;
      new_xyz[((size_t)b * NPOINT + it) * 3 + 1] = cy;
      new_xyz[((size_t)b * NPOINT + it) * 3 + 2] = cz;
    }
    float bv = -1.0f; int bi = 0x7fffffff;
#pragma unroll
    for (int j = 0; j < PTS; ++j) {
      int i = threadIdx.x + j * 256;
      float dx = X[i * 3 + 0] - cx;
      float dy = X[i * 3 + 1] - cy;
      float dz = X[i * 3 + 2] - cz;
      float d  = dx * dx + dy * dy + dz * dz;
      float nd = fminf(dist[j], d);
      dist[j] = nd;
      if (nd > bv) { bv = nd; bi = i; }  // ascending i -> first max (tie rule)
    }
    s_rv[threadIdx.x] = bv; s_ri[threadIdx.x] = bi;
    __syncthreads();
    for (int s = 128; s > 0; s >>= 1) {
      if ((int)threadIdx.x < s) {
        float v1 = s_rv[threadIdx.x];     int i1 = s_ri[threadIdx.x];
        float v2 = s_rv[threadIdx.x + s]; int i2 = s_ri[threadIdx.x + s];
        if (v2 > v1 || (v2 == v1 && i2 < i1)) {
          s_rv[threadIdx.x] = v2; s_ri[threadIdx.x] = i2;
        }
      }
      __syncthreads();
    }
    if (threadIdx.x == 0) s_far = s_ri[0];
    __syncthreads();
  }
}

// --- Kernel 2: kNN (k=16) via register insertion sort, one thread/centroid --
template<int N, int K>
__global__ __launch_bounds__(256)
void knn_kernel(const float* __restrict__ query, const float* __restrict__ ref,
                int* __restrict__ idx, int P, int total) {
  int t = blockIdx.x * 256 + threadIdx.x;
  if (t >= total) return;
  int b = t / P, p = t % P;
  const float* Q = query + ((size_t)b * P + p) * 3;
  const float* R = ref + (size_t)b * N * 3;
  float qx = Q[0], qy = Q[1], qz = Q[2];
  float dl[K]; int il[K];
#pragma unroll
  for (int j = 0; j < K; ++j) { dl[j] = 3.0e38f; il[j] = 0; }
  for (int i = 0; i < N; ++i) {
    float dx = R[i * 3 + 0] - qx;
    float dy = R[i * 3 + 1] - qy;
    float dz = R[i * 3 + 2] - qz;
    float d = dx * dx + dy * dy + dz * dz;
    if (d < dl[K - 1]) {               // strict < reproduces top_k tie order
      float vd = d; int vi = i;
#pragma unroll
      for (int j = 0; j < K; ++j) {
        if (vd < dl[j]) {
          float td = dl[j]; int ti = il[j];
          dl[j] = vd; il[j] = vi; vd = td; vi = ti;
        }
      }
    }
  }
  int* O = idx + ((size_t)b * P + p) * K;
#pragma unroll
  for (int j = 0; j < K; ++j) O[j] = il[j];
}

// --- Kernel 3: set-abstraction MLP (2 layers + max over K=16), WMMA ---------
template<int CIN, int KPAD, int C1, int C2, int NREF>
__global__ __launch_bounds__(128)
void sa_mlp_kernel(const float* __restrict__ xyz,      // (B,NREF,3)
                   const float* __restrict__ pts,      // (B,NREF,CIN-3)
                   const float* __restrict__ new_xyz,  // (B,P,3)
                   const int*   __restrict__ idx,      // (B,P,16)
                   const _Float16* __restrict__ pWa, const float* __restrict__ ba,
                   const _Float16* __restrict__ pWb, const float* __restrict__ bb,
                   float* __restrict__ out,            // (B,P,C2)
                   int P) {
  constexpr int CPTS  = CIN - 3;
  constexpr int WAVES = 4;
  constexpr int NT1   = C1 / 16;
  constexpr int NT2   = C2 / 16;
  __shared__ __align__(16) _Float16 s_feat[WAVES][16][KPAD];
  __shared__ __align__(16) _Float16 s_h[WAVES][16][C1];

  int wave = threadIdx.x >> 5;
  int lane = threadIdx.x & 31;
  int gp = blockIdx.x * WAVES + wave;   // grid sized so gp < B*P always
  int b = gp / P, p = gp % P;

  const int* nbr = idx + ((size_t)b * P + p) * 16;
  float cx = new_xyz[((size_t)b * P + p) * 3 + 0];
  float cy = new_xyz[((size_t)b * P + p) * 3 + 1];
  float cz = new_xyz[((size_t)b * P + p) * 3 + 2];

  // Build grouped feature tile: [centered xyz | gathered feats | zero pad]
  for (int e = lane; e < 16 * KPAD; e += 32) {
    int r = e / KPAD, c = e % KPAD;
    int j = nbr[r];
    float v = 0.0f;
    if (c < 3) {
      float ctr = (c == 0) ? cx : ((c == 1) ? cy : cz);
      v = xyz[((size_t)b * NREF + j) * 3 + c] - ctr;
    } else if (c < CIN) {
      v = pts[((size_t)b * NREF + j) * CPTS + (c - 3)];
    }
    s_feat[wave][r][c] = (_Float16)v;
  }
  __syncthreads();

  int col = lane & 15;
  int hi  = lane >> 4;

  // Layer 1: (16 x KPAD) @ (KPAD x C1), bias + ReLU -> LDS f16
#pragma unroll
  for (int nt = 0; nt < NT1; ++nt) {
    v8f c = {};
#pragma unroll
    for (int kt = 0; kt < KPAD / 32; ++kt) {
      v16h a  = load_a_frag(&s_feat[wave][0][kt * 32], KPAD);
      v16h bf = load_b_frag_packed(pWa + (size_t)(kt * NT1 + nt) * 512);
      c = wmma_f16(a, bf, c);
    }
    float bias = ba[nt * 16 + col];
#pragma unroll
    for (int r = 0; r < 8; ++r) {
      float v = fmaxf(c[r] + bias, 0.0f);
      s_h[wave][r + hi * 8][nt * 16 + col] = (_Float16)v;
    }
  }
  __syncthreads();

  // Layer 2: (16 x C1) @ (C1 x C2), then max over the 16 neighbor rows
#pragma unroll
  for (int nt = 0; nt < NT2; ++nt) {
    v8f c = {};
#pragma unroll
    for (int kt = 0; kt < C1 / 32; ++kt) {
      v16h a  = load_a_frag(&s_h[wave][0][kt * 32], C1);
      v16h bf = load_b_frag_packed(pWb + (size_t)(kt * NT2 + nt) * 512);
      c = wmma_f16(a, bf, c);
    }
    float m = c[0];
#pragma unroll
    for (int r = 1; r < 8; ++r) m = fmaxf(m, c[r]);
    m = fmaxf(m, __shfl_xor(m, 16, 32));  // fold the two M-halves (wave32)
    if (lane < 16)
      out[((size_t)b * P + p) * C2 + nt * 16 + col] = m + bb[nt * 16 + col];
  }
}

// --- Kernel 4: SA3 layer 1: (B,128,259) -> ReLU h3 (B,128,512) f16 ---------
__global__ __launch_bounds__(128)
void sa3_layer1_kernel(const float* __restrict__ l2_xyz,       // (B,128,3)
                       const float* __restrict__ l2_pts,       // (B,128,256)
                       const _Float16* __restrict__ pW3a,      // packed, NT=32
                       const float* __restrict__ b3a,
                       _Float16* __restrict__ h3) {            // (B,128,512)
  constexpr int CIN = 259, KPAD = 288, NT = 32;
  __shared__ __align__(16) _Float16 s_feat[16][KPAD];
  int b = blockIdx.x >> 3, rt = blockIdx.x & 7;
  int wave = threadIdx.x >> 5, lane = threadIdx.x & 31;

  for (int e = threadIdx.x; e < 16 * KPAD; e += 128) {
    int r = e / KPAD, c = e % KPAD;
    int n = rt * 16 + r;
    float v = 0.0f;
    if (c < 3)        v = l2_xyz[((size_t)b * 128 + n) * 3 + c];
    else if (c < CIN) v = l2_pts[((size_t)b * 128 + n) * 256 + (c - 3)];
    s_feat[r][c] = (_Float16)v;
  }
  __syncthreads();

  int col = lane & 15, hi = lane >> 4;
  for (int nt = wave; nt < NT; nt += 4) {
    v8f c = {};
#pragma unroll
    for (int kt = 0; kt < KPAD / 32; ++kt) {
      v16h a  = load_a_frag(&s_feat[0][kt * 32], KPAD);
      v16h bf = load_b_frag_packed(pW3a + (size_t)(kt * NT + nt) * 512);
      c = wmma_f16(a, bf, c);
    }
    float bias = b3a[nt * 16 + col];
#pragma unroll
    for (int r = 0; r < 8; ++r) {
      float v = fmaxf(c[r] + bias, 0.0f);
      h3[((size_t)b * 128 + rt * 16 + r + hi * 8) * 512 + nt * 16 + col] = (_Float16)v;
    }
  }
}

// --- Kernel 5: SA3 layer 2 + max over all 128 points -> (B,256) -------------
__global__ __launch_bounds__(256)
void sa3_layer2_kernel(const _Float16* __restrict__ h3,    // (B,128,512)
                       const _Float16* __restrict__ pW3b,  // packed, NT=16
                       const float* __restrict__ b3b,
                       float* __restrict__ out) {          // (B,256)
  constexpr int NT = 16;
  int b = blockIdx.x >> 4, nt = blockIdx.x & 15;
  int wave = threadIdx.x >> 5, lane = threadIdx.x & 31;
  __shared__ float s_max[8][16];

  const _Float16* A = h3 + ((size_t)b * 128 + wave * 16) * 512;
  v8f c = {};
#pragma unroll
  for (int kt = 0; kt < 16; ++kt) {
    v16h a  = load_a_frag(A + kt * 32, 512);
    v16h bf = load_b_frag_packed(pW3b + (size_t)(kt * NT + nt) * 512);
    c = wmma_f16(a, bf, c);
  }
  int col = lane & 15;
  float m = c[0];
#pragma unroll
  for (int r = 1; r < 8; ++r) m = fmaxf(m, c[r]);
  m = fmaxf(m, __shfl_xor(m, 16, 32));
  if (lane < 16) s_max[wave][col] = m;
  __syncthreads();
  if (threadIdx.x < 16) {
    float v = s_max[0][threadIdx.x];
#pragma unroll
    for (int w = 1; w < 8; ++w) v = fmaxf(v, s_max[w][threadIdx.x]);
    out[(size_t)b * 256 + nt * 16 + threadIdx.x] = v + b3b[nt * 16 + threadIdx.x];
  }
}

// ---------------------------------------------------------------------------
extern "C" void kernel_launch(void* const* d_in, const int* in_sizes, int n_in,
                              void* d_out, int out_size, void* d_ws, size_t ws_size,
                              hipStream_t stream) {
  (void)n_in; (void)out_size; (void)ws_size;
  const float* pc  = (const float*)d_in[0];
  const float* W1a = (const float*)d_in[1];  const float* b1a = (const float*)d_in[2];
  const float* W1b = (const float*)d_in[3];  const float* b1b = (const float*)d_in[4];
  const float* W2a = (const float*)d_in[5];  const float* b2a = (const float*)d_in[6];
  const float* W2b = (const float*)d_in[7];  const float* b2b = (const float*)d_in[8];
  const float* W3a = (const float*)d_in[9];  const float* b3a = (const float*)d_in[10];
  const float* W3b = (const float*)d_in[11]; const float* b3b = (const float*)d_in[12];

  const int N1 = 8192, P1 = 512, P2 = 128;
  const int B = in_sizes[0] / (3 * N1);

  // Workspace layout (256-B aligned regions; ~11.5 MB for B=16)
  char* ws = (char*)d_ws;
  float*     xyz1   = (float*)(ws + 0);             // B*8192*3 f32
  float*     nxyz1  = (float*)(ws + 1572864);       // B*512*3  f32
  int*       idx1   = (int*)  (ws + 1671168);       // B*512*16 i32
  float*     l1pts  = (float*)(ws + 2195456);       // B*512*128 f32
  float*     nxyz2  = (float*)(ws + 6389760);       // B*128*3  f32
  int*       idx2   = (int*)  (ws + 6414336);       // B*128*16 i32
  float*     l2pts  = (float*)(ws + 6545408);       // B*128*256 f32
  _Float16*  h3     = (_Float16*)(ws + 8642560);    // B*128*512 f16
  _Float16*  pW1a   = (_Float16*)(ws + 10739712);   //   4 KB (1 kt x 4 nt)
  _Float16*  pW1b   = (_Float16*)(ws + 10743808);   //  16 KB (2 x 8)
  _Float16*  pW2a   = (_Float16*)(ws + 10760192);   //  40 KB (5 x 8)
  _Float16*  pW2b   = (_Float16*)(ws + 10801152);   //  64 KB (4 x 16)
  _Float16*  pW3a   = (_Float16*)(ws + 10866688);   // 288 KB (9 x 32)
  _Float16*  pW3b   = (_Float16*)(ws + 11161600);   // 256 KB (16 x 16)

  // Pack all weights into B-fragment layout (f16), once per launch.
  pack_w_kernel<<<(1 * 4 * 512 + 255) / 256, 256, 0, stream>>>(W1a, pW1a, 6,   4,  1 * 4 * 512);
  pack_w_kernel<<<(2 * 8 * 512 + 255) / 256, 256, 0, stream>>>(W1b, pW1b, 64,  8,  2 * 8 * 512);
  pack_w_kernel<<<(5 * 8 * 512 + 255) / 256, 256, 0, stream>>>(W2a, pW2a, 131, 8,  5 * 8 * 512);
  pack_w_kernel<<<(4 * 16 * 512 + 255) / 256, 256, 0, stream>>>(W2b, pW2b, 128, 16, 4 * 16 * 512);
  pack_w_kernel<<<(9 * 32 * 512 + 255) / 256, 256, 0, stream>>>(W3a, pW3a, 259, 32, 9 * 32 * 512);
  pack_w_kernel<<<(16 * 16 * 512 + 255) / 256, 256, 0, stream>>>(W3b, pW3b, 512, 16, 16 * 16 * 512);

  // Stage 0: transpose point cloud to point-major
  transpose_pc_kernel<<<(B * N1 + 255) / 256, 256, 0, stream>>>(pc, xyz1, B, N1);

  // SA1: FPS 8192->512, kNN(16), MLP 6->64->128 (points == xyz, CPTS=3)
  fps_kernel<8192, 512><<<B, 256, 0, stream>>>(xyz1, nxyz1);
  knn_kernel<8192, 16><<<(B * P1 + 255) / 256, 256, 0, stream>>>(
      nxyz1, xyz1, idx1, P1, B * P1);
  sa_mlp_kernel<6, 32, 64, 128, 8192><<<(B * P1) / 4, 128, 0, stream>>>(
      xyz1, xyz1, nxyz1, idx1, pW1a, b1a, pW1b, b1b, l1pts, P1);

  // SA2: FPS 512->128, kNN(16), MLP 131->128->256
  fps_kernel<512, 128><<<B, 256, 0, stream>>>(nxyz1, nxyz2);
  knn_kernel<512, 16><<<(B * P2 + 255) / 256, 256, 0, stream>>>(
      nxyz2, nxyz1, idx2, P2, B * P2);
  sa_mlp_kernel<131, 160, 128, 256, 512><<<(B * P2) / 4, 128, 0, stream>>>(
      nxyz1, l1pts, nxyz2, idx2, pW2a, b2a, pW2b, b2b, l2pts, P2);

  // SA3 (group_all): MLP 259->512 (ReLU), 512->256, max over 128 points
  sa3_layer1_kernel<<<B * 8, 128, 0, stream>>>(nxyz2, l2pts, pW3a, b3a, h3);
  sa3_layer2_kernel<<<B * 16, 256, 0, stream>>>(h3, pW3b, b3b, (float*)d_out);
}
